// ODEFunc_48627619725834
// MI455X (gfx1250) — 16-bit and 8-bit matrix operand order repaired and checked by reading, compiled, not run on hardware
//
#include <hip/hip_runtime.h>


typedef __attribute__((ext_vector_type(16))) _Float16 v16h;
typedef __attribute__((ext_vector_type(8)))  float    v8f;
typedef __attribute__((ext_vector_type(8)))  int      v8i;

#define BB 2
#define NN 2048
#define DD 1024
#define HH 16
#define HDIM 64
#define MROWS (BB * NN)
#define K3 (3 * DD)
#define EPSF 1e-5f

union FragI { v8i v; unsigned u[8]; };
union FragH { v16h v; unsigned u[8]; };

__global__ void __launch_bounds__(256)
k_ln_rows(const float* __restrict__ x, const float* __restrict__ gamma,
          const float* __restrict__ beta, float* __restrict__ h,
          float* __restrict__ rsum, float* __restrict__ rss) {
  __shared__ float s1[256], s2[256];
  const int tid = threadIdx.x;
  const int row = blockIdx.x;
  const float* xr = x + (size_t)row * DD;
  float v[4];
  float a = 0.f, b = 0.f;
#pragma unroll
  for (int i = 0; i < 4; ++i) {
    v[i] = xr[tid + 256 * i];
    a += v[i]; b += v[i] * v[i];
  }
  s1[tid] = a; s2[tid] = b; __syncthreads();
  for (int off = 128; off > 0; off >>= 1) {
    if (tid < off) { s1[tid] += s1[tid + off]; s2[tid] += s2[tid + off]; }
    __syncthreads();
  }
  const float mean = s1[0] * (1.f / DD);
  const float var  = s2[0] * (1.f / DD) - mean * mean;
  const float rstd = rsqrtf(var + EPSF);
  __syncthreads();
  float ha = 0.f, hb = 0.f;
#pragma unroll
  for (int i = 0; i < 4; ++i) {
    const int col = tid + 256 * i;
    const float hv = (v[i] - mean) * rstd * gamma[col] + beta[col];
    h[(size_t)row * DD + col] = hv;
    ha += hv; hb += hv * hv;
  }
  s1[tid] = ha; s2[tid] = hb; __syncthreads();
  for (int off = 128; off > 0; off >>= 1) {
    if (tid < off) { s1[tid] += s1[tid + off]; s2[tid] += s2[tid + off]; }
    __syncthreads();
  }
  if (tid == 0) { rsum[row] = s1[0]; rss[row] = s2[0]; }
}

__global__ void __launch_bounds__(256)
k_reduce_pairs(const float* __restrict__ pa, const float* __restrict__ pb,
               int per_batch, float* __restrict__ acc) {
  __shared__ float s1[256], s2[256];
  const int tid = threadIdx.x, b = blockIdx.x;
  float a = 0.f, c = 0.f;
  for (int i = tid; i < per_batch; i += 256) {
    a += pa[(size_t)b * per_batch + i];
    c += pb[(size_t)b * per_batch + i];
  }
  s1[tid] = a; s2[tid] = c; __syncthreads();
  for (int off = 128; off > 0; off >>= 1) {
    if (tid < off) { s1[tid] += s1[tid + off]; s2[tid] += s2[tid + off]; }
    __syncthreads();
  }
  if (tid == 0) { acc[b * 2] = s1[0]; acc[b * 2 + 1] = s2[0]; }
}

__global__ void __launch_bounds__(256)
k_abs_part(const float* __restrict__ w, int n, float* __restrict__ part) {
  __shared__ float s1[256];
  const int tid = threadIdx.x;
  float a = 0.f;
  for (int i = blockIdx.x * 256 + tid; i < n; i += gridDim.x * 256) a += fabsf(w[i]);
  s1[tid] = a; __syncthreads();
  for (int off = 128; off > 0; off >>= 1) {
    if (tid < off) s1[tid] += s1[tid + off];
    __syncthreads();
  }
  if (tid == 0) part[blockIdx.x] = s1[0];
}

__global__ void __launch_bounds__(256)
k_reduce_scalar(const float* __restrict__ part, int n, float* __restrict__ out) {
  __shared__ float s1[256];
  const int tid = threadIdx.x;
  float a = 0.f;
  for (int i = tid; i < n; i += 256) a += part[i];
  s1[tid] = a; __syncthreads();
  for (int off = 128; off > 0; off >>= 1) {
    if (tid < off) s1[tid] += s1[tid + off];
    __syncthreads();
  }
  if (tid == 0) out[0] = s1[0];
}

__global__ void __launch_bounds__(256)
k_wquant(const float* __restrict__ w, int n, const float* __restrict__ abssum,
         signed char* __restrict__ wq) {
  const int i = blockIdx.x * 256 + threadIdx.x;
  if (i >= n) return;
  const float mean = fmaxf(abssum[0] / (float)n, EPSF);
  const float s = 1.f / mean;
  const float q = fminf(fmaxf(rintf(w[i] * s), -1.f), 1.f);
  wq[i] = (signed char)(int)q;
}

__global__ void __launch_bounds__(256)
k_actquant(const float* __restrict__ src, const float* __restrict__ bacc,
           signed char* __restrict__ qact, float* __restrict__ ascale) {
  __shared__ float s1[256];
  const int tid = threadIdx.x, row = blockIdx.x, b = row >> 11;
  const float inv_nd = 1.f / (float)(NN * DD);
  const float mean = bacc[b * 2] * inv_nd;
  const float var  = bacc[b * 2 + 1] * inv_nd - mean * mean;
  const float rstd = rsqrtf(var + EPSF);
  const float* sr = src + (size_t)row * DD;
  float v[4];
  float mx = 0.f;
#pragma unroll
  for (int i = 0; i < 4; ++i) {
    v[i] = (sr[tid + 256 * i] - mean) * rstd;
    mx = fmaxf(mx, fabsf(v[i]));
  }
  s1[tid] = mx; __syncthreads();
  for (int off = 128; off > 0; off >>= 1) {
    if (tid < off) s1[tid] = fmaxf(s1[tid], s1[tid + off]);
    __syncthreads();
  }
  const float amax = fmaxf(s1[0], EPSF);
  const float scale = 127.f / amax;
#pragma unroll
  for (int i = 0; i < 4; ++i) {
    const float q = fminf(fmaxf(rintf(v[i] * scale), -128.f), 127.f);
    qact[(size_t)row * DD + tid + 256 * i] = (signed char)(int)q;
  }
  if (tid == 0) ascale[row] = amax * (1.f / 127.f);
}

__device__ inline v8i load_fragA_i8(const signed char* base, int lane) {
  FragI f;
  const int m = lane & 15, half = lane >> 4;
#pragma unroll
  for (int v = 0; v < 8; ++v) {
    const int kb = ((v >> 1) << 4) + ((v & 1) << 2) + (half << 3);
    f.u[v] = *(const unsigned*)(base + (size_t)m * DD + kb);
  }
  return f.v;
}

__device__ inline v8i load_fragB_i8(const signed char* base, int lane) {
  FragI f;
  const int n = lane & 15, half = lane >> 4;
#pragma unroll
  for (int v = 0; v < 8; ++v) {
    const int kb = ((v >> 1) << 4) + ((v & 1) << 2) + (half << 3);
    f.u[v] = *(const unsigned*)(base + (size_t)n * DD + kb);
  }
  return f.v;
}

__global__ void __launch_bounds__(256)
k_gemm_qkv(const signed char* __restrict__ qa, const float* __restrict__ ascale,
           const signed char* __restrict__ qw, const float* __restrict__ wabs,
           _Float16* __restrict__ qh, _Float16* __restrict__ kh,
           _Float16* __restrict__ vh) {
  const int lane = threadIdx.x & 31, w = threadIdx.x >> 5;
  const int rowbase = blockIdx.y * 128 + w * 16;
  const int colbase = blockIdx.x * 32;
  v8i acc0 = {}, acc1 = {};
  for (int kk = 0; kk < DD; kk += 64) {
    const v8i a  = load_fragA_i8(qa + (size_t)rowbase * DD + kk, lane);
    const v8i b0 = load_fragB_i8(qw + (size_t)colbase * DD + kk, lane);
    const v8i b1 = load_fragB_i8(qw + (size_t)(colbase + 16) * DD + kk, lane);
    acc0 = __builtin_amdgcn_wmma_i32_16x16x64_iu8(true, a, true, b0, acc0, false, false);
    acc1 = __builtin_amdgcn_wmma_i32_16x16x64_iu8(true, a, true, b1, acc1, false, false);
  }
  const float wsc = fmaxf(wabs[0] * (1.f / (float)(K3 * DD)), EPSF);
  const int half = lane >> 4, n = lane & 15;
#pragma unroll
  for (int e = 0; e < 8; ++e) {
    const int rowm = rowbase + e + 8 * half;
    const float as = ascale[rowm] * wsc;
    const int b = rowm >> 11, nrow = rowm & (NN - 1);
#pragma unroll
    for (int t = 0; t < 2; ++t) {
      const int c = colbase + n + 16 * t;
      const int ai = (t == 0) ? acc0[e] : acc1[e];
      const float f = (float)ai * as;
      const int which = c >> 10, rem = c & 1023;
      const int hh = rem >> 6, dd = rem & 63;
      const size_t idx = (((size_t)b * HH + hh) * NN + nrow) * HDIM + dd;
      const _Float16 hv = (_Float16)f;
      if (which == 0) qh[idx] = hv;
      else if (which == 1) kh[idx] = hv;
      else vh[idx] = hv;
    }
  }
}

__global__ void __launch_bounds__(256)
k_gemm_out(const signed char* __restrict__ qa, const float* __restrict__ ascale,
           const signed char* __restrict__ qw, const float* __restrict__ wabs,
           float* __restrict__ out) {
  const int lane = threadIdx.x & 31, w = threadIdx.x >> 5;
  const int rowbase = blockIdx.y * 128 + w * 16;
  const int colbase = blockIdx.x * 32;
  v8i acc0 = {}, acc1 = {};
  for (int kk = 0; kk < DD; kk += 64) {
    const v8i a  = load_fragA_i8(qa + (size_t)rowbase * DD + kk, lane);
    const v8i b0 = load_fragB_i8(qw + (size_t)colbase * DD + kk, lane);
    const v8i b1 = load_fragB_i8(qw + (size_t)(colbase + 16) * DD + kk, lane);
    acc0 = __builtin_amdgcn_wmma_i32_16x16x64_iu8(true, a, true, b0, acc0, false, false);
    acc1 = __builtin_amdgcn_wmma_i32_16x16x64_iu8(true, a, true, b1, acc1, false, false);
  }
  const float wsc = fmaxf(wabs[0] * (1.f / (float)(DD * DD)), EPSF);
  const int half = lane >> 4, n = lane & 15;
#pragma unroll
  for (int e = 0; e < 8; ++e) {
    const int rowm = rowbase + e + 8 * half;
    const float as = ascale[rowm] * wsc;
    out[(size_t)rowm * DD + colbase + n]      = (float)acc0[e] * as;
    out[(size_t)rowm * DD + colbase + 16 + n] = (float)acc1[e] * as;
  }
}

__global__ void __launch_bounds__(256)
k_attn(const _Float16* __restrict__ qg, const _Float16* __restrict__ kg,
       const _Float16* __restrict__ vg, float* __restrict__ out,
       float* __restrict__ psum, float* __restrict__ pss) {
  __shared__ _Float16 Ksh[32 * 64];
  __shared__ _Float16 Vsh[64 * 34];
  __shared__ _Float16 Psh[8 * 16 * 34];
  const int tid = threadIdx.x;
  const int lane = tid & 31, w = tid >> 5;
  const int qblk = blockIdx.x & 15, bhid = blockIdx.x >> 4;
  const int h = bhid & 15, b = bhid >> 4;
  const size_t headoff = (size_t)bhid * NN * HDIM;
  const _Float16* Q = qg + headoff;
  const _Float16* K = kg + headoff;
  const _Float16* V = vg + headoff;
  const int m = lane & 15, half = lane >> 4;
  const int qbase = qblk * 128 + w * 16;

  FragH aQ[2];
#pragma unroll
  for (int c = 0; c < 2; ++c)
#pragma unroll
    for (int v = 0; v < 8; ++v) {
      const int kq = ((v >> 2) << 4) + (half << 3) + ((v & 3) << 1);
      aQ[c].u[v] = *(const unsigned*)(Q + (size_t)(qbase + m) * HDIM + c * 32 + kq);
    }

  float runmax[8], runsum[8];
  v8f O[4] = {};
#pragma unroll
  for (int e = 0; e < 8; ++e) { runmax[e] = -1e30f; runsum[e] = 0.f; }

  for (int kt = 0; kt < NN; kt += 32) {
    __syncthreads();
    {
      const unsigned* src = (const unsigned*)(K + (size_t)kt * HDIM);
      unsigned* dst = (unsigned*)Ksh;
#pragma unroll
      for (int i = 0; i < 4; ++i) dst[tid + 256 * i] = src[tid + 256 * i];
    }
    for (int e = tid; e < 32 * 64; e += 256) {
      const int key = e >> 6, dim = e & 63;
      Vsh[dim * 34 + key] = V[(size_t)(kt + key) * HDIM + dim];
    }
    __syncthreads();

    FragH bK[2][2];
#pragma unroll
    for (int nt = 0; nt < 2; ++nt)
#pragma unroll
      for (int c = 0; c < 2; ++c)
#pragma unroll
        for (int v = 0; v < 8; ++v) {
          const int d = c * 32 + ((v >> 2) << 4) + (half << 3) + ((v & 3) << 1);
          bK[nt][c].u[v] = *(const unsigned*)(Ksh + (nt * 16 + m) * 64 + d);
        }
    v8f s0 = {}, s1 = {};
    s0 = __builtin_amdgcn_wmma_f32_16x16x32_f16(false, aQ[0].v, false, bK[0][0].v, (short)0, s0, false, false);
    s0 = __builtin_amdgcn_wmma_f32_16x16x32_f16(false, aQ[1].v, false, bK[0][1].v, (short)0, s0, false, false);
    s1 = __builtin_amdgcn_wmma_f32_16x16x32_f16(false, aQ[0].v, false, bK[1][0].v, (short)0, s1, false, false);
    s1 = __builtin_amdgcn_wmma_f32_16x16x32_f16(false, aQ[1].v, false, bK[1][1].v, (short)0, s1, false, false);

    _Float16* Pw = Psh + w * 16 * 34;
#pragma unroll
    for (int e = 0; e < 8; ++e) {
      float a0 = s0[e] * 0.125f;
      float a1 = s1[e] * 0.125f;
      float tm = fmaxf(a0, a1);
#pragma unroll
      for (int off = 1; off < 16; off <<= 1) tm = fmaxf(tm, __shfl_xor(tm, off, 32));
      const float nm = fmaxf(runmax[e], tm);
      const float alpha = __expf(runmax[e] - nm);
      runmax[e] = nm;
      const float p0 = __expf(a0 - nm), p1 = __expf(a1 - nm);
      float rs = p0 + p1;
#pragma unroll
      for (int off = 1; off < 16; off <<= 1) rs += __shfl_xor(rs, off, 32);
      runsum[e] = runsum[e] * alpha + rs;
#pragma unroll
      for (int t = 0; t < 4; ++t) O[t][e] = O[t][e] * alpha;
      const int mr = e + 8 * half;
      Pw[mr * 34 + m]      = (_Float16)p0;
      Pw[mr * 34 + m + 16] = (_Float16)p1;
    }

    FragH aP;
#pragma unroll
    for (int v = 0; v < 8; ++v) {
      const int kp = ((v >> 2) << 4) + (half << 3) + ((v & 3) << 1);
      aP.u[v] = *(const unsigned*)(Pw + m * 34 + kp);
    }
#pragma unroll
    for (int t = 0; t < 4; ++t) {
      FragH bV;
#pragma unroll
      for (int v = 0; v < 8; ++v) {
        const int key = ((v >> 2) << 4) + (half << 3) + ((v & 3) << 1);
        bV.u[v] = *(const unsigned*)(Vsh + (t * 16 + m) * 34 + key);
      }
      O[t] = __builtin_amdgcn_wmma_f32_16x16x32_f16(false, aP.v, false, bV.v, (short)0, O[t], false, false);
    }
  }

  float ls = 0.f, lss = 0.f;
#pragma unroll
  for (int t = 0; t < 4; ++t)
#pragma unroll
    for (int e = 0; e < 8; ++e) {
      const int mr = e + 8 * half;
      const float val = O[t][e] / runsum[e];
      const int nrow = qbase + mr;
      const int dg = h * HDIM + t * 16 + m;
      out[((size_t)b * NN + nrow) * DD + dg] = val;
      ls += val; lss += val * val;
    }
#pragma unroll
  for (int off = 1; off < 32; off <<= 1) {
    ls  += __shfl_xor(ls, off, 32);
    lss += __shfl_xor(lss, off, 32);
  }
  if (lane == 0) {
    psum[(size_t)blockIdx.x * 8 + w] = ls;
    pss[(size_t)blockIdx.x * 8 + w]  = lss;
  }
}

extern "C" void kernel_launch(void* const* d_in, const int* in_sizes, int n_in,
                              void* d_out, int out_size, void* d_ws, size_t ws_size,
                              hipStream_t stream) {
  (void)in_sizes; (void)n_in; (void)out_size; (void)ws_size;
  const float* x     = (const float*)d_in[1];
  const float* w_qkv = (const float*)d_in[2];
  const float* w_out = (const float*)d_in[3];
  const float* gamma = (const float*)d_in[4];
  const float* beta  = (const float*)d_in[5];
  float* outp = (float*)d_out;

  char* ws = (char*)d_ws;
  size_t off = 0;
  auto alloc = [&](size_t bytes) { size_t cur = off; off += (bytes + 255) & ~(size_t)255; return cur; };

  const size_t o_acc   = alloc(16 * sizeof(float));
  const size_t o_red1  = alloc((size_t)MROWS * sizeof(float));
  const size_t o_red2  = alloc((size_t)MROWS * sizeof(float));
  const size_t o_wpart = alloc(512 * sizeof(float));
  const size_t o_bigA  = alloc((size_t)MROWS * DD * sizeof(float));
  const size_t o_act   = alloc((size_t)MROWS * DD);
  const size_t o_ascl  = alloc((size_t)MROWS * sizeof(float));
  const size_t o_wqkv  = alloc((size_t)K3 * DD);
  const size_t o_wout  = alloc((size_t)DD * DD);
  const size_t o_v     = alloc((size_t)BB * HH * NN * HDIM * sizeof(_Float16));

  float* acc   = (float*)(ws + o_acc);
  float* red1  = (float*)(ws + o_red1);
  float* red2  = (float*)(ws + o_red2);
  float* wpq   = (float*)(ws + o_wpart);
  float* wpo   = wpq + 256;
  float* hbuf  = (float*)(ws + o_bigA);
  signed char* qact = (signed char*)(ws + o_act);
  float* ascl  = (float*)(ws + o_ascl);
  signed char* qwqkv = (signed char*)(ws + o_wqkv);
  signed char* qwout = (signed char*)(ws + o_wout);
  _Float16* qh = (_Float16*)(ws + o_bigA);
  _Float16* kh = qh + (size_t)BB * HH * NN * HDIM;
  _Float16* vh = (_Float16*)(ws + o_v);

  k_ln_rows<<<MROWS, 256, 0, stream>>>(x, gamma, beta, hbuf, red1, red2);
  k_reduce_pairs<<<BB, 256, 0, stream>>>(red1, red2, NN, acc + 0);

  k_abs_part<<<256, 256, 0, stream>>>(w_qkv, K3 * DD, wpq);
  k_reduce_scalar<<<1, 256, 0, stream>>>(wpq, 256, acc + 8);
  k_abs_part<<<256, 256, 0, stream>>>(w_out, DD * DD, wpo);
  k_reduce_scalar<<<1, 256, 0, stream>>>(wpo, 256, acc + 9);
  k_wquant<<<(K3 * DD + 255) / 256, 256, 0, stream>>>(w_qkv, K3 * DD, acc + 8, qwqkv);
  k_wquant<<<(DD * DD + 255) / 256, 256, 0, stream>>>(w_out, DD * DD, acc + 9, qwout);

  k_actquant<<<MROWS, 256, 0, stream>>>(hbuf, acc + 0, qact, ascl);
  k_gemm_qkv<<<dim3(K3 / 32, MROWS / 128), 256, 0, stream>>>(qact, ascl, qwqkv, acc + 8, qh, kh, vh);

  k_attn<<<BB * HH * (NN / 128), 256, 0, stream>>>(qh, kh, vh, outp, red1, red2);
  k_reduce_pairs<<<BB, 256, 0, stream>>>(red1, red2, NN, acc + 4);

  k_actquant<<<MROWS, 256, 0, stream>>>(outp, acc + 4, qact, ascl);
  k_gemm_out<<<dim3(DD / 32, MROWS / 128), 256, 0, stream>>>(qact, ascl, qwout, acc + 9, outp);
}
